// GCNPolicyNetwork_17214228923074
// MI455X (gfx1250) — matrix operand order repaired and checked by reading, not compiled, not run on hardware
//
#include <hip/hip_runtime.h>
#include <hip/hip_bf16.h>

typedef __attribute__((ext_vector_type(16))) __bf16 v16bf;
typedef __attribute__((ext_vector_type(8)))  float  v8f;

#define N_NODES 50000
#define N_EDGES 800000
#define N_GRAPHS 128
#define D_IN 128
#define D_HID 128
#define N_ACT 64

// ---------------- generic fill ----------------
__global__ void fill_f32(float* __restrict__ p, float v, long n) {
    long i  = (long)blockIdx.x * blockDim.x + threadIdx.x;
    long st = (long)gridDim.x * blockDim.x;
    for (; i < n; i += st) p[i] = v;
}

// ---------------- degree / norm ----------------
__global__ void deg_accum(const int* __restrict__ dst, float* __restrict__ deg, int nE) {
    int i = blockIdx.x * blockDim.x + threadIdx.x;
    if (i < nE) atomicAdd(&deg[dst[i]], 1.0f);
}

__global__ void make_dinv(float* __restrict__ deg, int n) {
    int i = blockIdx.x * blockDim.x + threadIdx.x;
    if (i < n) {
        float d = deg[i];
        deg[i] = (d > 0.0f) ? rsqrtf(d) : 0.0f;
    }
}

// ---------------- B-fragment prep: f32 [K x N] -> WMMA bf16 frag order ----------------
// frag layout: [kt][nt][lane(32)][16 bf16], lane: col = nt*16 + (lane&15),
// element e -> K = kt*32 + (lane>=16)*16 + e   (contiguous k order, same as the A frag in gemm_wmma)
__global__ void prep_bfrag(const float* __restrict__ W, __bf16* __restrict__ F, int N, int K) {
    int lane = threadIdx.x & 31;
    int wid  = (blockIdx.x * blockDim.x + threadIdx.x) >> 5;
    int ntiles = N >> 4, ktiles = K >> 5;
    if (wid >= ktiles * ntiles) return;
    int kt = wid / ntiles, nt = wid - kt * ntiles;
    int half = lane >> 4;
    int col  = nt * 16 + (lane & 15);
    __bf16* out = F + (((size_t)wid * 32) + lane) * 16;
#pragma unroll
    for (int e = 0; e < 16; e++) {
        int k = kt * 32 + half * 16 + e;
        out[e] = (__bf16)W[(size_t)k * N + col];
    }
}

// ---------------- WMMA GEMM: C[M x (NT*16)] = A[M x 128](f32->bf16) * Bfrag (+bias) ----
template <int NT>
__global__ void gemm_wmma(const float* __restrict__ A, const __bf16* __restrict__ BF,
                          const float* __restrict__ bias, float* __restrict__ C, int M) {
    constexpr int N = NT * 16;
    constexpr int K = 128;
    int lane = threadIdx.x & 31;
    int wv   = threadIdx.x >> 5;
    int r0   = (blockIdx.x * 8 + wv) * 16;
    if (r0 >= M) return;                     // uniform per-wave exit keeps EXEC full
    int half = lane >> 4;
    int ra   = r0 + (lane & 15);
    if (ra > M - 1) ra = M - 1;              // clamp reads; writes guarded below
    const float* arow = A + (size_t)ra * K;

    v8f acc[NT];
#pragma unroll
    for (int i = 0; i < NT; i++)
#pragma unroll
        for (int j = 0; j < 8; j++) acc[i][j] = 0.0f;

#pragma unroll
    for (int kt = 0; kt < 4; kt++) {
        // A frag: contiguous k order per lane (k = half*16 + e), matching Bfrag
        int k0 = kt * 32 + half * 16;
        v16bf a;
#pragma unroll
        for (int e = 0; e < 8; e++) {
            a[e]     = (__bf16)arow[k0 + e];
            a[e + 8] = (__bf16)arow[k0 + 8 + e];
        }
        const v16bf* bp = (const v16bf*)(BF + (((size_t)kt * NT) * 32 + lane) * 16);
#pragma unroll
        for (int nt = 0; nt < NT; nt++) {
            v16bf b = bp[nt * 32];           // 32 v16bf per tile (32 lanes)
            acc[nt] = __builtin_amdgcn_wmma_f32_16x16x32_bf16(
                false, a, false, b, (short)0, acc[nt], false, false);
        }
    }

    int colbase = lane & 15;
#pragma unroll
    for (int nt = 0; nt < NT; nt++) {
        int col  = nt * 16 + colbase;
        float bb = bias ? bias[col] : 0.0f;
#pragma unroll
        for (int v = 0; v < 8; v++) {
            int row = r0 + half * 8 + v;     // C/D layout: VGPR v -> M = v + half*8
            if (row < M) C[(size_t)row * N + col] = acc[nt][v] + bb;
        }
    }
}

// ---------------- edge gather/scatter: one wave per edge ----------------
__global__ void aggregate_edges(const float* __restrict__ T, const float* __restrict__ dinv,
                                const int* __restrict__ src, const int* __restrict__ dst,
                                float* __restrict__ OUT, int nE) {
    int lane = threadIdx.x & 31;
    int wid  = (blockIdx.x * blockDim.x + threadIdx.x) >> 5;
    if (wid >= nE) return;
    int s = src[wid], d = dst[wid];
    float nrm = dinv[s] * dinv[d];
    float4 v = ((const float4*)(T + (size_t)s * D_HID))[lane];
    float* o = OUT + (size_t)d * D_HID + lane * 4;
    atomicAdd(o + 0, v.x * nrm);
    atomicAdd(o + 1, v.y * nrm);
    atomicAdd(o + 2, v.z * nrm);
    atomicAdd(o + 3, v.w * nrm);
}

// ---------------- self-loop + bias + (relu), in place on AGG ----------------
__global__ void finish_layer(float* __restrict__ AGG, const float* __restrict__ T,
                             const float* __restrict__ dinv, const float* __restrict__ bias,
                             int n, int doRelu) {
    long tid   = (long)blockIdx.x * blockDim.x + threadIdx.x;
    long total = (long)n * (D_HID / 4);
    if (tid >= total) return;
    int node = (int)(tid >> 5);
    int f4   = (int)(tid & 31);
    float di = dinv[node];
    float s  = di * di;                      // self-loop norm = dinv[i]*dinv[i]
    float4 a = ((float4*)AGG)[tid];
    float4 t = ((const float4*)T)[tid];
    float4 b = ((const float4*)bias)[f4];
    float4 r;
    r.x = a.x + t.x * s + b.x;
    r.y = a.y + t.y * s + b.y;
    r.z = a.z + t.z * s + b.z;
    r.w = a.w + t.w * s + b.w;
    if (doRelu) {
        r.x = fmaxf(r.x, 0.0f); r.y = fmaxf(r.y, 0.0f);
        r.z = fmaxf(r.z, 0.0f); r.w = fmaxf(r.w, 0.0f);
    }
    ((float4*)AGG)[tid] = r;
}

// ---------------- mean pool ----------------
__global__ void pool_accum(const float* __restrict__ H, const int* __restrict__ batch,
                           float* __restrict__ pooled, float* __restrict__ cnt, int n) {
    int lane = threadIdx.x & 31;
    int wid  = (blockIdx.x * blockDim.x + threadIdx.x) >> 5;
    if (wid >= n) return;
    int g = batch[wid];
    float4 v = ((const float4*)(H + (size_t)wid * D_HID))[lane];
    float* o = pooled + (size_t)g * D_HID + lane * 4;
    atomicAdd(o + 0, v.x);
    atomicAdd(o + 1, v.y);
    atomicAdd(o + 2, v.z);
    atomicAdd(o + 3, v.w);
    if (lane == 0) atomicAdd(&cnt[g], 1.0f);
}

__global__ void pool_div(float* __restrict__ pooled, const float* __restrict__ cnt) {
    int i = blockIdx.x * blockDim.x + threadIdx.x;
    if (i < N_GRAPHS * D_HID) pooled[i] /= fmaxf(cnt[i >> 7], 1.0f);
}

// ---------------- launch ----------------
extern "C" void kernel_launch(void* const* d_in, const int* in_sizes, int n_in,
                              void* d_out, int out_size, void* d_ws, size_t ws_size,
                              hipStream_t stream) {
    const float* x      = (const float*)d_in[0];
    const int*   eidx   = (const int*)d_in[1];   // [2][E]
    const int*   batch  = (const int*)d_in[2];
    const float* W1     = (const float*)d_in[3];
    const float* b1     = (const float*)d_in[4];
    const float* W2     = (const float*)d_in[5];
    const float* b2     = (const float*)d_in[6];
    const float* Wh     = (const float*)d_in[7];
    const float* bh     = (const float*)d_in[8];
    float*       out    = (float*)d_out;
    const int* esrc = eidx;
    const int* edst = eidx + N_EDGES;

    // workspace carve-up
    char* w = (char*)d_ws;
    float* T    = (float*)w;  w += (size_t)N_NODES * D_HID * 4;   // 25.6 MB
    float* AG   = (float*)w;  w += (size_t)N_NODES * D_HID * 4;   // 25.6 MB
    float* dinv = (float*)w;  w += ((N_NODES * 4 + 255) / 256) * 256;
    __bf16* f1  = (__bf16*)w; w += D_IN  * D_HID * 2;
    __bf16* f2  = (__bf16*)w; w += D_HID * D_HID * 2;
    __bf16* fh  = (__bf16*)w; w += D_HID * N_ACT * 2;
    float* pooled = (float*)w; w += N_GRAPHS * D_HID * 4;
    float* cnt    = (float*)w; w += ((N_GRAPHS * 4 + 255) / 256) * 256;

    const long nodeFeat = (long)N_NODES * D_HID;

    // 1) degrees (init 1.0 = self-loop) -> dinv in place
    fill_f32<<<256, 256, 0, stream>>>(dinv, 1.0f, N_NODES);
    deg_accum<<<(N_EDGES + 255) / 256, 256, 0, stream>>>(edst, dinv, N_EDGES);
    make_dinv<<<(N_NODES + 255) / 256, 256, 0, stream>>>(dinv, N_NODES);

    // 2) weight fragments (bf16, WMMA B order)
    prep_bfrag<<<4, 256, 0, stream>>>(W1, f1, D_HID, D_IN);   // 32 waves
    prep_bfrag<<<4, 256, 0, stream>>>(W2, f2, D_HID, D_HID);
    prep_bfrag<<<2, 256, 0, stream>>>(Wh, fh, N_ACT, D_HID);  // 16 waves

    const int gemmGrid = (N_NODES + 127) / 128;
    const int edgeGrid = (N_EDGES + 7) / 8;       // 8 waves / block
    const int elemGrid = (int)((nodeFeat / 4 + 255) / 256);

    // layer 1
    gemm_wmma<8><<<gemmGrid, 256, 0, stream>>>(x, f1, nullptr, T, N_NODES);
    fill_f32<<<2048, 256, 0, stream>>>(AG, 0.0f, nodeFeat);
    aggregate_edges<<<edgeGrid, 256, 0, stream>>>(T, dinv, esrc, edst, AG, N_EDGES);
    finish_layer<<<elemGrid, 256, 0, stream>>>(AG, T, dinv, b1, N_NODES, 1);

    // layer 2
    gemm_wmma<8><<<gemmGrid, 256, 0, stream>>>(AG, f2, nullptr, T, N_NODES);
    fill_f32<<<2048, 256, 0, stream>>>(AG, 0.0f, nodeFeat);
    aggregate_edges<<<edgeGrid, 256, 0, stream>>>(T, dinv, esrc, edst, AG, N_EDGES);
    finish_layer<<<elemGrid, 256, 0, stream>>>(AG, T, dinv, b2, N_NODES, 0);

    // pooling (pooled & cnt contiguous -> one fill)
    fill_f32<<<64, 256, 0, stream>>>(pooled, 0.0f, N_GRAPHS * D_HID + N_GRAPHS);
    pool_accum<<<(N_NODES + 7) / 8, 256, 0, stream>>>(AG, batch, pooled, cnt, N_NODES);
    pool_div<<<(N_GRAPHS * D_HID + 255) / 256, 256, 0, stream>>>(pooled, cnt);

    // head: 128 x 64, bias, straight to d_out
    gemm_wmma<4><<<1, 256, 0, stream>>>(pooled, fh, bh, out, N_GRAPHS);
}
